// TripletLoss_23716809408616
// MI455X (gfx1250) — matrix operand order repaired and checked by reading, not compiled, not run on hardware
//
#include <hip/hip_runtime.h>

typedef __attribute__((ext_vector_type(2)))  unsigned int v2u;
typedef __attribute__((ext_vector_type(4)))  unsigned int v4u;
typedef __attribute__((ext_vector_type(8)))  unsigned int v8u;
typedef __attribute__((ext_vector_type(4)))  float v4f;
typedef __attribute__((ext_vector_type(8)))  float v8f;
typedef __attribute__((ext_vector_type(16))) __bf16 v16bf;

#define NROWS  128
#define KDIM   131072
#define SPLITK 128                 // # partial-sum blocks (grid for gram kernel)
#define KTOT   (KDIM / SPLITK)     // 1024 K-columns per block
#define KC     64                  // K-columns per double-buffered stage
#define NSTAGE (KTOT / KC)         // 16 stages; acc lives in registers throughout
#define STRIDE 144                 // padded LDS row stride B (72 bf16): 36 dwords,
                                   // gcd(36,64)=4 -> 16 rows hit 16 disjoint 4-bank runs
#define PLANE  (NROWS * STRIDE)    // 18432 B per bf16 plane
#define LDSBUF (2 * PLANE)         // hi|lo planes = one stage buffer (36864 B)
#define GELEMS (NROWS * NROWS)
#define NREG   8                   // float4 prefetch regs per thread (256*8*4 = 128*64)

// round-to-nearest-even f32 -> bf16 (top 16 bits)
__device__ __forceinline__ unsigned bf16_rne(float x) {
  unsigned u = __float_as_uint(x);
  return (u + 0x7FFFu + ((u >> 16) & 1u)) >> 16;
}

// assemble a 16-element bf16 fragment from two 16B LDS chunks
__device__ __forceinline__ v16bf ld_frag(const unsigned char* s, int o0, int o1) {
  v4u q0 = *(const v4u*)(s + o0);
  v4u q1 = *(const v4u*)(s + o1);
  v8u q  = __builtin_shufflevector(q0, q1, 0, 1, 2, 3, 4, 5, 6, 7);
  return __builtin_bit_cast(v16bf, q);
}

// ---------------------------------------------------------------------------
// Split-K Gram partials with double-buffered LDS staging:
//   P[b] = X[:, b*KTOT : (b+1)*KTOT] * (same)^T
// f32 is split once per element into bf16 hi+lo; G accumulated as
// hi*hi + hi*lo + lo*hi via v_wmma_f32_16x16x32_bf16 (fp32 accumulators).
// Loop pipeline per stage: convert prefetched regs -> LDS buf[st&1], barrier,
// issue global prefetch for stage st+1, WMMA-compute stage st from LDS.
// ---------------------------------------------------------------------------
__global__ __launch_bounds__(256) void gram_bf16_splitk(const float* __restrict__ X,
                                                        float* __restrict__ P) {
  __shared__ __align__(16) unsigned char smem[2 * LDSBUF];

  const int t    = threadIdx.x;
  const int lane = t & 31;
  const int wave = t >> 5;
  const int twy  = wave & 3;       // tile-row pair
  const int twx  = wave >> 2;      // tile-col quad
  const int lr   = lane & 15;
  const int lh   = lane >> 4;      // lane half selects K sub-range (ISA 7.12.2)

  // staging geometry: thread t covers (row = v>>4, kk = (v&15)*4), v = t + 256*i
  const int kblock = blockIdx.x * KTOT;
  size_t gbase[NREG];
  int    soff[NREG];
#pragma unroll
  for (int i = 0; i < NREG; ++i) {
    int v   = t + 256 * i;
    int row = v >> 4;
    int kk  = (v & 15) << 2;
    gbase[i] = (size_t)row * KDIM + (kblock + kk);
    soff[i]  = row * STRIDE + (kk << 1);
  }

  v8f acc[2][4] = {};
  v4f regs[NREG];

  // prologue: prefetch stage 0
#pragma unroll
  for (int i = 0; i < NREG; ++i) regs[i] = *(const v4f*)(X + gbase[i]);

  for (int st = 0; st < NSTAGE; ++st) {
    unsigned char* buf = smem + (st & 1) * LDSBUF;

    // ---- convert prefetched f32 -> (hi,lo) bf16 -> LDS ----
#pragma unroll
    for (int i = 0; i < NREG; ++i) {
      v4f x = regs[i];
      unsigned h0 = bf16_rne(x.x), h1 = bf16_rne(x.y);
      unsigned h2 = bf16_rne(x.z), h3 = bf16_rne(x.w);
      float r0 = x.x - __uint_as_float(h0 << 16);
      float r1 = x.y - __uint_as_float(h1 << 16);
      float r2 = x.z - __uint_as_float(h2 << 16);
      float r3 = x.w - __uint_as_float(h3 << 16);
      v2u hi, lo;
      hi.x = h0 | (h1 << 16);
      hi.y = h2 | (h3 << 16);
      lo.x = bf16_rne(r0) | (bf16_rne(r1) << 16);
      lo.y = bf16_rne(r2) | (bf16_rne(r3) << 16);
      *(v2u*)(buf + soff[i])         = hi;
      *(v2u*)(buf + PLANE + soff[i]) = lo;
    }
    __syncthreads();   // stores visible; also fences WAR on the other buffer

    // ---- prefetch next stage while this stage computes ----
    if (st + 1 < NSTAGE) {
      const int koff = (st + 1) * KC;
#pragma unroll
      for (int i = 0; i < NREG; ++i) regs[i] = *(const v4f*)(X + gbase[i] + koff);
    }

    // ---- compute: KC/32 K-steps of bf16 WMMA from LDS ----
#pragma unroll
    for (int ks = 0; ks < KC / 32; ++ks) {
      const int kb = ks * 64;      // byte offset of K-slice (32 cols * 2B)
      v16bf ahi[2], alo[2];
#pragma unroll
      for (int r = 0; r < 2; ++r) {
        int ab = ((twy * 2 + r) * 16 + lr) * STRIDE + kb + lh * 16;
        ahi[r] = ld_frag(buf,         ab, ab + 32);
        alo[r] = ld_frag(buf + PLANE, ab, ab + 32);
      }
#pragma unroll
      for (int j = 0; j < 4; ++j) {
        int bb = ((twx * 4 + j) * 16 + lr) * STRIDE + kb + lh * 16;
        v16bf bhi = ld_frag(buf,         bb, bb + 32);
        v16bf blo = ld_frag(buf + PLANE, bb, bb + 32);
#pragma unroll
        for (int r = 0; r < 2; ++r) {
          acc[r][j] = __builtin_amdgcn_wmma_f32_16x16x32_bf16(
              false, ahi[r], false, bhi, (short)0, acc[r][j], false, false);
          acc[r][j] = __builtin_amdgcn_wmma_f32_16x16x32_bf16(
              false, ahi[r], false, blo, (short)0, acc[r][j], false, false);
          acc[r][j] = __builtin_amdgcn_wmma_f32_16x16x32_bf16(
              false, alo[r], false, bhi, (short)0, acc[r][j], false, false);
        }
      }
    }
  }

  // ---- write partial tiles (plain stores; no atomics) ----
  // C/D layout: VGPR v -> (M = v + 8*lane[4], N = lane&15)
  float* Pb = P + (size_t)blockIdx.x * GELEMS;
#pragma unroll
  for (int r = 0; r < 2; ++r) {
    const int gi0 = (twy * 2 + r) * 16 + 8 * lh;
#pragma unroll
    for (int j = 0; j < 4; ++j) {
      const int gj = (twx * 4 + j) * 16 + lr;
#pragma unroll
      for (int v = 0; v < 8; ++v)
        Pb[(gi0 + v) * NROWS + gj] = acc[r][j][v];
    }
  }
}

// ---------------------------------------------------------------------------
// Fold SPLITK partial Gram matrices into the final G (coalesced).
// ---------------------------------------------------------------------------
__global__ __launch_bounds__(256) void reduce_partials(const float* __restrict__ P,
                                                       float* __restrict__ G) {
  const int e = blockIdx.x * 256 + threadIdx.x;
  float s = 0.0f;
#pragma unroll 8
  for (int b = 0; b < SPLITK; ++b) s += P[(size_t)b * GELEMS + e];
  G[e] = s;
}

// ---------------------------------------------------------------------------
// Triplet margin loss from the Gram matrix. One block, 128 lanes (one per row).
// dist_ij = sqrt(max(G_ii + G_jj - 2 G_ij, EPS)); cross-modality masks;
// row-wise hard-positive max / hard-negative min; mean of relu(ap-an+margin).
// ---------------------------------------------------------------------------
__global__ __launch_bounds__(128) void triplet_loss(const float* __restrict__ G,
                                                    const long long* __restrict__ tgt,
                                                    float* __restrict__ out) {
  __shared__ float sq[NROWS];
  __shared__ long long lab[NROWS];
  __shared__ float red[NROWS];

  const int i = threadIdx.x;
  sq[i]  = G[i * NROWS + i];
  lab[i] = tgt[i];
  __syncthreads();

  const float sqi    = sq[i];
  const long long ti = lab[i];
  const bool halfi   = (i >= NROWS / 2);

  float ap = -__builtin_inff();
  float an =  __builtin_inff();
  for (int j = 0; j < NROWS; ++j) {
    const bool cross = ((j >= NROWS / 2) != halfi);
    if (cross) {
      float d2 = sqi + sq[j] - 2.0f * G[i * NROWS + j];
      float d  = sqrtf(fmaxf(d2, 1e-12f));
      if (lab[j] == ti) ap = fmaxf(ap, d);
      else              an = fminf(an, d);
    }
  }

  red[i] = fmaxf(ap - an + 0.3f, 0.0f);
  __syncthreads();
#pragma unroll
  for (int s = NROWS / 2; s > 0; s >>= 1) {
    if (i < s) red[i] += red[i + s];
    __syncthreads();
  }
  if (i == 0) out[0] = red[0] * (1.0f / NROWS);
}

// ---------------------------------------------------------------------------
extern "C" void kernel_launch(void* const* d_in, const int* in_sizes, int n_in,
                              void* d_out, int out_size, void* d_ws, size_t ws_size,
                              hipStream_t stream) {
  const float*     X   = (const float*)d_in[0];      // (128, 131072) f32
  const long long* tgt = (const long long*)d_in[1];  // (128,) int64
  float*           out = (float*)d_out;              // scalar f32

  float* P = (float*)d_ws;                           // SPLITK * 128*128 partials
  float* G = P + (size_t)SPLITK * GELEMS;            // final Gram (64 KB)

  gram_bf16_splitk<<<SPLITK, 256, 0, stream>>>(X, P);
  reduce_partials<<<GELEMS / 256, 256, 0, stream>>>(P, G);
  triplet_loss<<<1, NROWS, 0, stream>>>(G, tgt, out);
}
